// CatGWRegression_64149631533694
// MI455X (gfx1250) — matrix operand order repaired and checked by reading, not compiled, not run on hardware
//
#include <hip/hip_runtime.h>
#include <hip/hip_bf16.h>
#include <math.h>

#define N_NODES 4096
#define FDIM    128
#define HEADS   4
#define FOUT    64
#define D2      256
#define NVAR    16
#define H1DIM   96
#define H2DIM   32

typedef __attribute__((ext_vector_type(16))) _Float16 v16h;
typedef __attribute__((ext_vector_type(8)))  _Float16 v8h;
typedef __attribute__((ext_vector_type(8)))  float    v8f;
typedef __attribute__((ext_vector_type(4)))  float    v4f;

static __device__ __forceinline__ v16h cat8(v8h lo, v8h hi) {
  v16h r;
#pragma unroll
  for (int q = 0; q < 8; ++q) { r[q] = lo[q]; r[q + 8] = hi[q]; }
  return r;
}

static __device__ __forceinline__ v8f wmma16(v16h a, v16h b, v8f c) {
  // v_wmma_f32_16x16x32_f16: D = A(16x32 f16) * B(32x16 f16) + C(16x16 f32)
  return __builtin_amdgcn_wmma_f32_16x16x32_f16(false, a, false, b, (short)0, c,
                                                false, false);
}

// ---------------------------------------------------------------------------
// K0: wsrc = Wp2 @ a_src2, wtgt = Wp2 @ a_tgt2   (256-vectors)
// ---------------------------------------------------------------------------
__global__ void k_wvec(const float* __restrict__ Wp2, const float* __restrict__ a_src2,
                       const float* __restrict__ a_tgt2, float* __restrict__ wsrc,
                       float* __restrict__ wtgt) {
  int k = threadIdx.x;  // 256 threads, 1 block
  const float* row = Wp2 + (size_t)k * D2;
  float s = 0.f, t = 0.f;
  for (int c = 0; c < D2; ++c) { float w = row[c]; s += w * a_src2[c]; t += w * a_tgt2[c]; }
  wsrc[k] = s; wtgt[k] = t;
}

// ---------------------------------------------------------------------------
// K1: x (N x 128 f32) -> f16
// ---------------------------------------------------------------------------
__global__ void k_cvt_x(const float* __restrict__ x, _Float16* __restrict__ xh) {
  int i = blockIdx.x * 256 + threadIdx.x;
  if (i < N_NODES * FDIM) xh[i] = (_Float16)x[i];
}

// ---------------------------------------------------------------------------
// K2: WcatT (512 x 128 f16) = transpose of [Wp1 | Wskip1] (128 x 512)
// ---------------------------------------------------------------------------
__global__ void k_wcatT(const float* __restrict__ Wp1, const float* __restrict__ Wskip1,
                        _Float16* __restrict__ WcT) {
  int idx = blockIdx.x * 256 + threadIdx.x;  // 512*128
  if (idx >= 512 * FDIM) return;
  int c = idx / FDIM, k = idx % FDIM;
  float v = (c < D2) ? Wp1[(size_t)k * D2 + c] : Wskip1[(size_t)k * D2 + (c - D2)];
  WcT[idx] = (_Float16)v;
}

// ---------------------------------------------------------------------------
// K3: PS (N x 512 f32) = x_f16 @ [Wp1|Wskip1]_f16   (WMMA)
//   grid = N/16 blocks, 4 waves/block; wave w owns cols [w*128, w*128+128)
// ---------------------------------------------------------------------------
__global__ void k_gemm1(const _Float16* __restrict__ xh, const _Float16* __restrict__ WcT,
                        float* __restrict__ PS) {
  const int lane = threadIdx.x & 31;
  const int wave = threadIdx.x >> 5;
  const int m0   = blockIdx.x * 16;
  const int lm   = lane & 15;
  const int off8   = (lane & 16) ? 8 : 0;
  const int koff16 = (lane & 16) ? 16 : 0;
  const int rbase  = (lane & 16) ? 8 : 0;
  v8f acc[8] = {};
  const _Float16* xrow = xh + (size_t)(m0 + lm) * FDIM;
#pragma unroll
  for (int kt = 0; kt < FDIM; kt += 32) {
    v8h a0 = *(const v8h*)(xrow + kt + off8);
    v8h a1 = *(const v8h*)(xrow + kt + off8 + 16);
    v16h A = cat8(a0, a1);
#pragma unroll
    for (int t = 0; t < 8; ++t) {
      int c = wave * 128 + t * 16 + lm;
      const _Float16* wr = WcT + (size_t)c * FDIM + kt + off8;
      v16h B = cat8(*(const v8h*)wr, *(const v8h*)(wr + 16));
      acc[t] = wmma16(A, B, acc[t]);
    }
  }
#pragma unroll
  for (int t = 0; t < 8; ++t) {
    int c = wave * 128 + t * 16 + lm;
#pragma unroll
    for (int r = 0; r < 8; ++r)
      PS[(size_t)(m0 + rbase + r) * 512 + c] = acc[t][r];
  }
}

// ---------------------------------------------------------------------------
// K4: s_src[h,n] / s_tgt[h,n] from PS (p part = cols 0..255)
//   one wave per node n; lane covers 8 cols, 8-lane groups = one head
// ---------------------------------------------------------------------------
__global__ void k_attn_vec(const float* __restrict__ PS, const float* __restrict__ a_src1,
                           const float* __restrict__ a_tgt1, float* __restrict__ s_src,
                           float* __restrict__ s_tgt) {
  int lane = threadIdx.x & 31;
  int wave = threadIdx.x >> 5;
  int n = blockIdx.x * 4 + wave;
  const float* prow = PS + (size_t)n * 512;
  int c0 = lane * 8;
  float ps = 0.f, pt = 0.f;
#pragma unroll
  for (int q = 0; q < 8; ++q) {
    float p = prow[c0 + q];
    ps += p * a_src1[c0 + q];
    pt += p * a_tgt1[c0 + q];
  }
#pragma unroll
  for (int m = 1; m < 8; m <<= 1) { ps += __shfl_xor(ps, m); pt += __shfl_xor(pt, m); }
  if ((lane & 7) == 0) {
    int h = lane >> 3;
    s_src[h * N_NODES + n] = ps;
    s_tgt[h * N_NODES + n] = pt;
  }
}

// ---------------------------------------------------------------------------
// K5: PT (256 x 4096 f16) = transpose of PS[:, :256]  (L2-resident B matrix)
// ---------------------------------------------------------------------------
__global__ void k_PT(const float* __restrict__ PS, _Float16* __restrict__ PT) {
  int idx = blockIdx.x * 256 + threadIdx.x;  // 256*4096
  if (idx >= D2 * N_NODES) return;
  int c = idx >> 12, n = idx & 4095;
  PT[idx] = (_Float16)PS[(size_t)n * 512 + c];
}

// ---------------------------------------------------------------------------
// K6: flash-attention GAT layer. grid = N/16, block = 128 (wave == head).
//   out1[i, h*64+f] = elu( sum_j softmax_j(lrelu(ssrc+stgt)+mask) * P + skip + bias )
// ---------------------------------------------------------------------------
__global__ void k_attn(const float* __restrict__ sw, const float* __restrict__ s_src,
                       const float* __restrict__ s_tgt, const _Float16* __restrict__ PT,
                       const float* __restrict__ PS, const float* __restrict__ bias1,
                       float* __restrict__ out1) {
  const int lane = threadIdx.x & 31;
  const int h    = threadIdx.x >> 5;
  const int m0   = blockIdx.x * 16;
  const int lm   = lane & 15;
  const int off8   = (lane & 16) ? 8 : 0;
  const int koff16 = (lane & 16) ? 16 : 0;
  const int rbase  = (lane & 16) ? 8 : 0;
  const int i = m0 + lm;
  const float ssrc_i = s_src[h * N_NODES + i];
  const float* swrow = sw + (size_t)i * N_NODES;
  const float* strow = s_tgt + h * N_NODES;
  float m_run = -1e30f, l_run = 0.f;
  v8f acc[4] = {};
  for (int jt = 0; jt < N_NODES; jt += 32) {
    const int b0 = jt + off8;
    float s[16];
#pragma unroll
    for (int ch = 0; ch < 2; ++ch) {               // A-layout K chunks: +0 and +16
      int base = b0 + ch * 16;
      v4f swa = *(const v4f*)(swrow + base);
      v4f swb = *(const v4f*)(swrow + base + 4);
      v4f sta = *(const v4f*)(strow + base);
      v4f stb = *(const v4f*)(strow + base + 4);
#pragma unroll
      for (int q = 0; q < 4; ++q) {
        float z0 = ssrc_i + sta[q]; z0 = (z0 > 0.f) ? z0 : 0.2f * z0;
        s[ch * 8 + q]     = z0 + ((swa[q] > 0.f) ? 0.f : -1e30f);
        float z1 = ssrc_i + stb[q]; z1 = (z1 > 0.f) ? z1 : 0.2f * z1;
        s[ch * 8 + 4 + q] = z1 + ((swb[q] > 0.f) ? 0.f : -1e30f);
      }
    }
    float tm = s[0];
#pragma unroll
    for (int q = 1; q < 16; ++q) tm = fmaxf(tm, s[q]);
    tm = fmaxf(tm, __shfl_xor(tm, 16));            // row i lives in lanes i and i+16
    float m_new = fmaxf(m_run, tm);
    float scale = __expf(m_run - m_new);
    float p[16], psum = 0.f;
#pragma unroll
    for (int q = 0; q < 16; ++q) { p[q] = __expf(s[q] - m_new); psum += p[q]; }
    l_run = l_run * scale + psum + __shfl_xor(psum, 16);
    m_run = m_new;
#pragma unroll
    for (int r = 0; r < 8; ++r) {                  // rescale C by per-row factor
      float f = __shfl(scale, rbase + r);
#pragma unroll
      for (int t = 0; t < 4; ++t) acc[t][r] *= f;
    }
    v16h A;
#pragma unroll
    for (int q = 0; q < 16; ++q) A[q] = (_Float16)p[q];
#pragma unroll
    for (int t = 0; t < 4; ++t) {
      int c = h * 64 + t * 16 + lm;
      const _Float16* pr = PT + (size_t)c * N_NODES + jt + off8;
      v16h B = cat8(*(const v8h*)pr, *(const v8h*)(pr + 16));
      acc[t] = wmma16(A, B, acc[t]);
    }
  }
  float inv = 1.f / l_run;
#pragma unroll
  for (int r = 0; r < 8; ++r) {
    float f  = __shfl(inv, rbase + r);
    int  row = m0 + rbase + r;
#pragma unroll
    for (int t = 0; t < 4; ++t) {
      int col = h * 64 + t * 16 + lm;
      float o = acc[t][r] * f + PS[(size_t)row * 512 + D2 + col] + bias1[col];
      o = (o > 0.f) ? o : (__expf(o) - 1.f);       // elu
      out1[(size_t)row * D2 + col] = o;
    }
  }
}

// ---------------------------------------------------------------------------
// K7: ss[n] = out1[n,:] . wsrc ; st[n] = out1[n,:] . wtgt
// ---------------------------------------------------------------------------
__global__ void k_sst(const float* __restrict__ out1, const float* __restrict__ wsrc,
                      const float* __restrict__ wtgt, float* __restrict__ ssv,
                      float* __restrict__ stv) {
  int lane = threadIdx.x & 31;
  int wave = threadIdx.x >> 5;
  int n = blockIdx.x * 4 + wave;
  const float* row = out1 + (size_t)n * D2;
  float a = 0.f, b = 0.f;
  for (int c = lane; c < D2; c += 32) { float v = row[c]; a += v * wsrc[c]; b += v * wtgt[c]; }
#pragma unroll
  for (int m = 1; m < 32; m <<= 1) { a += __shfl_xor(a, m); b += __shfl_xor(b, m); }
  if (lane == 0) { ssv[n] = a; stv[n] = b; }
}

// ---------------------------------------------------------------------------
// K8: W1T (96 x 4096 f16) = transpose of W1 (4096 x 96)
// ---------------------------------------------------------------------------
__global__ void k_w1t(const float* __restrict__ W1, _Float16* __restrict__ W1T) {
  int idx = blockIdx.x * 256 + threadIdx.x;  // 96*4096
  if (idx >= H1DIM * N_NODES) return;
  int c = idx >> 12, k = idx & 4095;
  W1T[idx] = (_Float16)W1[(size_t)k * H1DIM + c];
}

// ---------------------------------------------------------------------------
// K9: fused  msw = relu(ss+st)*sw  (written to d_out)  and  t1 = msw@W1 + b1
//   grid = N/64 blocks, 4 waves/block, each wave: 16 rows x 96 cols (6 n-tiles)
// ---------------------------------------------------------------------------
__global__ void k_gemm2(const float* __restrict__ ssv, const float* __restrict__ stv,
                        const float* __restrict__ sw, const _Float16* __restrict__ W1T,
                        const float* __restrict__ rb1, float* __restrict__ msw_out,
                        float* __restrict__ t1) {
  const int lane = threadIdx.x & 31;
  const int wave = threadIdx.x >> 5;
  const int m0   = blockIdx.x * 64 + wave * 16;
  const int lm   = lane & 15;
  const int off8   = (lane & 16) ? 8 : 0;
  const int koff16 = (lane & 16) ? 16 : 0;
  const int rbase  = (lane & 16) ? 8 : 0;
  const int i = m0 + lm;
  const float ss_i = ssv[i];
  const float* swrow = sw + (size_t)i * N_NODES;
  float* mrow = msw_out + (size_t)i * N_NODES;
  v8f acc[6] = {};
  for (int jt = 0; jt < N_NODES; jt += 32) {
    const int b0 = jt + off8;
    float mv[16];
#pragma unroll
    for (int ch = 0; ch < 2; ++ch) {
      int base = b0 + ch * 16;
      v4f sta = *(const v4f*)(stv + base);
      v4f stb = *(const v4f*)(stv + base + 4);
      v4f swa = *(const v4f*)(swrow + base);
      v4f swb = *(const v4f*)(swrow + base + 4);
      v4f oa, ob;
#pragma unroll
      for (int q = 0; q < 4; ++q) {
        float z = ss_i + sta[q]; z = (z > 0.f) ? z : 0.f;
        oa[q] = z * swa[q];
        float z2 = ss_i + stb[q]; z2 = (z2 > 0.f) ? z2 : 0.f;
        ob[q] = z2 * swb[q];
        mv[ch * 8 + q] = oa[q]; mv[ch * 8 + 4 + q] = ob[q];
      }
      *(v4f*)(mrow + base)     = oa;               // msw written exactly once
      *(v4f*)(mrow + base + 4) = ob;
    }
    v16h A;
#pragma unroll
    for (int q = 0; q < 16; ++q) A[q] = (_Float16)mv[q];
#pragma unroll
    for (int t = 0; t < 6; ++t) {
      const _Float16* wr = W1T + (size_t)(t * 16 + lm) * N_NODES + jt + off8;
      v16h B = cat8(*(const v8h*)wr, *(const v8h*)(wr + 16));
      acc[t] = wmma16(A, B, acc[t]);
    }
  }
#pragma unroll
  for (int t = 0; t < 6; ++t) {
    int col = t * 16 + lm;
#pragma unroll
    for (int r = 0; r < 8; ++r)
      t1[(size_t)(m0 + rbase + r) * H1DIM + col] = acc[t][r] + rb1[col];
  }
}

// ---------------------------------------------------------------------------
// K10: t2 = t1@W2+b2 ; t3 = t2@W3+b3 ; beta = lrelu(t3)*ols ; y_hat
//   one wave per row (4 waves/block)
// ---------------------------------------------------------------------------
__global__ void k_head(const float* __restrict__ t1, const float* __restrict__ W2,
                       const float* __restrict__ b2, const float* __restrict__ W3,
                       const float* __restrict__ b3, const float* __restrict__ ols,
                       const float* __restrict__ vx, float* __restrict__ beta_out,
                       float* __restrict__ yhat) {
  __shared__ float t2buf[4][H2DIM];
  int lane = threadIdx.x & 31;
  int wave = threadIdx.x >> 5;
  int row = blockIdx.x * 4 + wave;
  const float* trow = t1 + (size_t)row * H1DIM;
  float t2 = b2[lane];
  for (int k = 0; k < H1DIM; ++k) t2 += trow[k] * W2[k * H2DIM + lane];
  t2buf[wave][lane] = t2;
  __syncthreads();
  int c = lane & 15;
  float t3 = b3[c];
#pragma unroll
  for (int k = 0; k < H2DIM; ++k) t3 += t2buf[wave][k] * W3[k * NVAR + c];
  float bt = ((t3 > 0.f) ? t3 : 0.2f * t3) * ols[c];
  float prod = bt * vx[(size_t)row * NVAR + c];
#pragma unroll
  for (int m = 1; m < 16; m <<= 1) prod += __shfl_xor(prod, m);
  if (lane < 16) beta_out[(size_t)row * NVAR + c] = bt;
  if (lane == 0) yhat[row] = prod;
}

// ---------------------------------------------------------------------------
extern "C" void kernel_launch(void* const* d_in, const int* in_sizes, int n_in,
                              void* d_out, int out_size, void* d_ws, size_t ws_size,
                              hipStream_t stream) {
  (void)in_sizes; (void)n_in; (void)out_size; (void)ws_size;
  const float* x      = (const float*)d_in[0];
  // d_in[1] = edge_distances (unused by the reference)
  const float* sw     = (const float*)d_in[2];
  const float* Wp1    = (const float*)d_in[3];
  const float* a_src1 = (const float*)d_in[4];
  const float* a_tgt1 = (const float*)d_in[5];
  const float* Wskip1 = (const float*)d_in[6];
  const float* bias1  = (const float*)d_in[7];
  const float* Wp2    = (const float*)d_in[8];
  const float* a_src2 = (const float*)d_in[9];
  const float* a_tgt2 = (const float*)d_in[10];
  const float* W1     = (const float*)d_in[11];
  const float* b1     = (const float*)d_in[12];
  const float* W2     = (const float*)d_in[13];
  const float* b2     = (const float*)d_in[14];
  const float* W3     = (const float*)d_in[15];
  const float* b3     = (const float*)d_in[16];
  const float* ols    = (const float*)d_in[17];
  const float* vx     = (const float*)d_in[18];

  char* ws = (char*)d_ws;
  float*    PS    = (float*)   (ws + 0);          //  8 MB  (N x 512)
  _Float16* xh    = (_Float16*)(ws + 8388608);    //  1 MB
  _Float16* WcT   = (_Float16*)(ws + 9437184);    //  128 KB
  _Float16* PT    = (_Float16*)(ws + 9568256);    //  2 MB  (256 x N, L2-resident)
  float*    s_src = (float*)   (ws + 11665408);   //  64 KB
  float*    s_tgt = (float*)   (ws + 11730944);   //  64 KB
  float*    out1  = (float*)   (ws + 11796480);   //  4 MB  (N x 256)
  float*    ssv   = (float*)   (ws + 15990784);   //  16 KB
  float*    stv   = (float*)   (ws + 16007168);   //  16 KB
  _Float16* W1T   = (_Float16*)(ws + 16023552);   //  768 KB
  float*    t1    = (float*)   (ws + 16809984);   //  1.5 MB (N x 96)
  float*    wsrc  = (float*)   (ws + 18382848);   //  1 KB
  float*    wtgt  = (float*)   (ws + 18383872);   //  1 KB

  float* beta_out = (float*)d_out;                // N*V
  float* yhat     = (float*)d_out + 65536;        // N
  float* msw_out  = (float*)d_out + 69632;        // N*N

  k_wvec<<<1, 256, 0, stream>>>(Wp2, a_src2, a_tgt2, wsrc, wtgt);
  k_cvt_x<<<(N_NODES * FDIM + 255) / 256, 256, 0, stream>>>(x, xh);
  k_wcatT<<<(512 * FDIM + 255) / 256, 256, 0, stream>>>(Wp1, Wskip1, WcT);
  k_gemm1<<<N_NODES / 16, 128, 0, stream>>>(xh, WcT, PS);
  k_attn_vec<<<N_NODES / 4, 128, 0, stream>>>(PS, a_src1, a_tgt1, s_src, s_tgt);
  k_PT<<<(D2 * N_NODES + 255) / 256, 256, 0, stream>>>(PS, PT);
  k_attn<<<N_NODES / 16, 128, 0, stream>>>(sw, s_src, s_tgt, PT, PS, bias1, out1);
  k_sst<<<N_NODES / 4, 128, 0, stream>>>(out1, wsrc, wtgt, ssv, stv);
  k_w1t<<<(H1DIM * N_NODES + 255) / 256, 256, 0, stream>>>(W1, W1T);
  k_gemm2<<<N_NODES / 64, 128, 0, stream>>>(ssv, stv, sw, W1T, b1, msw_out, t1);
  k_head<<<N_NODES / 4, 128, 0, stream>>>(t1, W2, b2, W3, b3, ols, vx, beta_out, yhat);
}
